// MTRNN_20521353740318
// MI455X (gfx1250) — matrix operand order repaired and checked by reading, not compiled, not run on hardware
//
#include <hip/hip_runtime.h>
#include <hip/hip_bf16.h>

typedef __attribute__((ext_vector_type(16))) _Float16 v16h;
typedef __attribute__((ext_vector_type(8)))  _Float16 v8h;
typedef __attribute__((ext_vector_type(8)))  float    v8f;

#define B_ROWS 524288
#define IN_F   32
#define OUT_F  32
#define IO_F   64
#define CF_F   128

// Weight fragment layout in d_ws (units of _Float16 halves).
// One fragment = one 32x16 (KxN) f16 B-tile = 32 lanes * 16 halves.
#define FRAG_HALVES 512
#define OFF_I2IO    0                    // 4 frags  (N=64 -> 4 ntiles, K=32  -> 1 kstep)
#define OFF_IO2IO   (4  * FRAG_HALVES)   // 8 frags  (4 ntiles, 2 ksteps)
#define OFF_CF2IO   (12 * FRAG_HALVES)   // 16 frags (4 ntiles, 4 ksteps)
#define OFF_IO2O    (28 * FRAG_HALVES)   // 4 frags  (2 ntiles, 2 ksteps)
#define TOTAL_W_HALVES (32 * FRAG_HALVES)

// ---------------------------------------------------------------------------
// Prep kernel 1: repack f32 row-major weights [N][K] into WMMA B-fragment
// layout, converted to f16. B layout (16-bit, 32x16 KxN): lane holds column
// N = lane&15; halves 0-7 hold K = g*8 + j, halves 8-15 hold K = 16 + g*8 + j
// (g = lane>>4), matching the A-fragment K order.
// ---------------------------------------------------------------------------
__global__ void mtrnn_prep_weights(const float* __restrict__ w_i2io,
                                   const float* __restrict__ w_io2io,
                                   const float* __restrict__ w_cf2io,
                                   const float* __restrict__ w_io2o,
                                   _Float16* __restrict__ frags) {
  const int f    = blockIdx.x;    // 0..31 fragment id
  const int lane = threadIdx.x;   // 0..31
  const float* W; int Ktot, nt, s, base;
  if (f < 4)       { W = w_i2io;  Ktot = 32;  int r = f;      nt = r;      s = 0;     base = OFF_I2IO  + r * FRAG_HALVES; }
  else if (f < 12) { W = w_io2io; Ktot = 64;  int r = f - 4;  nt = r >> 1; s = r & 1; base = OFF_IO2IO + r * FRAG_HALVES; }
  else if (f < 28) { W = w_cf2io; Ktot = 128; int r = f - 12; nt = r >> 2; s = r & 3; base = OFF_CF2IO + r * FRAG_HALVES; }
  else             { W = w_io2o;  Ktot = 64;  int r = f - 28; nt = r >> 1; s = r & 1; base = OFF_IO2O  + r * FRAG_HALVES; }
  const int n = nt * 16 + (lane & 15);
  const int g = lane >> 4;
  _Float16* dst = frags + base + lane * 16;
#pragma unroll
  for (int j = 0; j < 16; ++j) {
    const int k = s * 32 + (j < 8 ? g * 8 + j : 16 + g * 8 + (j - 8));
    dst[j] = (_Float16)W[n * Ktot + k];
  }
}

// Prep kernel 2: fold the three io-path biases, pre-divided by TAU_IO = 2.
__global__ void mtrnn_prep_bias(const float* __restrict__ b_i2io,
                                const float* __restrict__ b_io2io,
                                const float* __restrict__ b_cf2io,
                                const float* __restrict__ b_io2o,
                                float* __restrict__ bias_io_half,
                                float* __restrict__ b_y) {
  const int t = threadIdx.x;
  if (t < 64) bias_io_half[t] = 0.5f * (b_i2io[t] + b_io2io[t] + b_cf2io[t]);
  if (t < 32) b_y[t] = b_io2o[t];
}

// ---------------------------------------------------------------------------
// Main fused kernel.
// ---------------------------------------------------------------------------
__device__ __forceinline__ float fast_tanh(float v) {
#if __has_builtin(__builtin_amdgcn_tanhf)
  return __builtin_amdgcn_tanhf(v);   // V_TANH_F32 (CDNA5 trans op)
#else
  return tanhf(v);
#endif
}

// Build a 16x32 f16 A-fragment for this lane from a row-major f32 row chunk
// (32 consecutive K values). A layout: lane holds row M=lane&15; halves 0-7
// hold K = g*8 + j, halves 8-15 hold K = 16 + g*8 + j  (g = lane>>4).
__device__ __forceinline__ v16h load_afrag(const float* __restrict__ chunk, int g) {
  const float4* p = reinterpret_cast<const float4*>(chunk);
  float4 f0 = p[2 * g + 0];
  float4 f1 = p[2 * g + 1];
  float4 f2 = p[4 + 2 * g + 0];
  float4 f3 = p[4 + 2 * g + 1];
  v16h a;
  a[0]  = (_Float16)f0.x; a[1]  = (_Float16)f0.y; a[2]  = (_Float16)f0.z; a[3]  = (_Float16)f0.w;
  a[4]  = (_Float16)f1.x; a[5]  = (_Float16)f1.y; a[6]  = (_Float16)f1.z; a[7]  = (_Float16)f1.w;
  a[8]  = (_Float16)f2.x; a[9]  = (_Float16)f2.y; a[10] = (_Float16)f2.z; a[11] = (_Float16)f2.w;
  a[12] = (_Float16)f3.x; a[13] = (_Float16)f3.y; a[14] = (_Float16)f3.z; a[15] = (_Float16)f3.w;
  return a;
}

__device__ __forceinline__ v16h load_bfrag(const _Float16* __restrict__ frags,
                                           int off_halves, int lane) {
  return *reinterpret_cast<const v16h*>(frags + off_halves + lane * 16);
}

__global__ __launch_bounds__(256) void mtrnn_fused(
    const float* __restrict__ x,   const float* __restrict__ io,
    const float* __restrict__ cf,  const _Float16* __restrict__ frags,
    const float* __restrict__ bias_io_half, const float* __restrict__ b_y,
    float* __restrict__ y) {
  __shared__ __align__(32) _Float16 lds_nio[8 * 16 * IO_F];  // 16 KB, per-wave 16x64 tile

  const int  lane = threadIdx.x & 31;
  const int  wave = threadIdx.x >> 5;
  const int  m    = lane & 15;   // A row / D column
  const int  g    = lane >> 4;
  const long r0   = (long)(blockIdx.x * 8 + wave) * 16;

  // --- A fragments for the io-path GEMM: K = [x(32) | io(64) | cf(128)] ---
  const v16h a_x   = load_afrag(x  + (r0 + m) * IN_F, g);
  const v16h a_io0 = load_afrag(io + (r0 + m) * IO_F + 0,  g);
  const v16h a_io1 = load_afrag(io + (r0 + m) * IO_F + 32, g);
  v16h a_cf[4];
#pragma unroll
  for (int s = 0; s < 4; ++s)
    a_cf[s] = load_afrag(cf + (r0 + m) * CF_F + s * 32, g);

  _Float16* my_lds = lds_nio + wave * (16 * IO_F);

  // --- new_io: 4 N-tiles x 7 K-steps of v_wmma_f32_16x16x32_f16 ---
#pragma unroll
  for (int t = 0; t < 4; ++t) {
    v8f acc = {};
    acc = __builtin_amdgcn_wmma_f32_16x16x32_f16(false, a_x,   false,
            load_bfrag(frags, OFF_I2IO  + t * FRAG_HALVES,           lane), (short)0, acc, false, false);
    acc = __builtin_amdgcn_wmma_f32_16x16x32_f16(false, a_io0, false,
            load_bfrag(frags, OFF_IO2IO + (t * 2 + 0) * FRAG_HALVES, lane), (short)0, acc, false, false);
    acc = __builtin_amdgcn_wmma_f32_16x16x32_f16(false, a_io1, false,
            load_bfrag(frags, OFF_IO2IO + (t * 2 + 1) * FRAG_HALVES, lane), (short)0, acc, false, false);
#pragma unroll
    for (int s = 0; s < 4; ++s)
      acc = __builtin_amdgcn_wmma_f32_16x16x32_f16(false, a_cf[s], false,
              load_bfrag(frags, OFF_CF2IO + (t * 4 + s) * FRAG_HALVES, lane), (short)0, acc, false, false);

    const float bias = bias_io_half[t * 16 + m];
    // D layout: lane holds column n = t*16+m, VGPR v holds row v + 8*g.
#pragma unroll
    for (int v = 0; v < 8; ++v) {
      const long row  = r0 + v + 8 * g;
      const float prev = io[row * IO_F + t * 16 + m];
      const float nio  = fast_tanh(acc[v] * 0.5f + bias + 0.5f * prev);
      my_lds[(v + 8 * g) * IO_F + t * 16 + m] = (_Float16)nio;
    }
  }

  __syncthreads();

  // --- re-read new_io from LDS in A-fragment layout ---
  v16h a_n[2];
#pragma unroll
  for (int s = 0; s < 2; ++s) {
    const _Float16* lrow = my_lds + m * IO_F + s * 32;
    v8h lo = *reinterpret_cast<const v8h*>(lrow + g * 8);
    v8h hi = *reinterpret_cast<const v8h*>(lrow + 16 + g * 8);
    v16h a;
#pragma unroll
    for (int j = 0; j < 8; ++j) { a[j] = lo[j]; a[8 + j] = hi[j]; }
    a_n[s] = a;
  }

  // --- y = tanh(new_io @ W_io2o^T + b): 2 N-tiles x 2 K-steps ---
#pragma unroll
  for (int t = 0; t < 2; ++t) {
    v8f acc = {};
    acc = __builtin_amdgcn_wmma_f32_16x16x32_f16(false, a_n[0], false,
            load_bfrag(frags, OFF_IO2O + (t * 2 + 0) * FRAG_HALVES, lane), (short)0, acc, false, false);
    acc = __builtin_amdgcn_wmma_f32_16x16x32_f16(false, a_n[1], false,
            load_bfrag(frags, OFF_IO2O + (t * 2 + 1) * FRAG_HALVES, lane), (short)0, acc, false, false);
    const float bb = b_y[t * 16 + m];
#pragma unroll
    for (int v = 0; v < 8; ++v) {
      const long row = r0 + v + 8 * g;
      y[row * OUT_F + t * 16 + m] = fast_tanh(acc[v] + bb);
    }
  }
}

// ---------------------------------------------------------------------------
extern "C" void kernel_launch(void* const* d_in, const int* in_sizes, int n_in,
                              void* d_out, int out_size, void* d_ws, size_t ws_size,
                              hipStream_t stream) {
  // setup_inputs order: x, io_state, cf_state, cs_state, then (w,b) pairs:
  // i2io, io2o, io2io, io2cf, cf2io, cf2cs, cf2cf, cs2cf, cs2cs
  const float* x       = (const float*)d_in[0];
  const float* io      = (const float*)d_in[1];
  const float* cf      = (const float*)d_in[2];
  const float* w_i2io  = (const float*)d_in[4];
  const float* b_i2io  = (const float*)d_in[5];
  const float* w_io2o  = (const float*)d_in[6];
  const float* b_io2o  = (const float*)d_in[7];
  const float* w_io2io = (const float*)d_in[8];
  const float* b_io2io = (const float*)d_in[9];
  const float* w_cf2io = (const float*)d_in[12];
  const float* b_cf2io = (const float*)d_in[13];

  _Float16* frags      = (_Float16*)d_ws;
  float* bias_io_half  = (float*)((char*)d_ws + (size_t)TOTAL_W_HALVES * 2);
  float* b_y           = bias_io_half + 64;

  mtrnn_prep_weights<<<32, 32, 0, stream>>>(w_i2io, w_io2io, w_cf2io, w_io2o, frags);
  mtrnn_prep_bias<<<1, 64, 0, stream>>>(b_i2io, b_io2io, b_cf2io, b_io2o, bias_io_half, b_y);

  // 8 waves/block, 16 rows/wave -> 128 rows/block -> 4096 blocks
  mtrnn_fused<<<B_ROWS / 128, 256, 0, stream>>>(x, io, cf, frags,
                                                bias_io_half, b_y, (float*)d_out);
}
